// MiniGAT_86388972191822
// MI455X (gfx1250) — matrix operand order repaired and checked by reading, not compiled, not run on hardware
//
#include <hip/hip_runtime.h>
#include <hip/hip_bf16.h>
#include <stdint.h>

// ---------------- problem constants (match reference) ----------------
#define N_NODES  50000
#define N_EDGES  800000
#define E_TOT    (N_EDGES + N_NODES)   // self-loops appended
#define N_GRAPHS 64
#define IN_DIM   773
#define KP_IN    800                   // IN_DIM padded to multiple of 32
#define HID      256
#define OUTD     128
#define HEADS    4
#define NEG_SLOPE 0.2f

typedef __attribute__((ext_vector_type(16))) __bf16 v16bf;
typedef __attribute__((ext_vector_type(8)))  float  v8f;

static __device__ __forceinline__ uint16_t f32_to_bf16(float f) {
    uint32_t x = __float_as_uint(f);
    uint32_t lsb = (x >> 16) & 1u;
    x += 0x7fffu + lsb;               // round-to-nearest-even
    return (uint16_t)(x >> 16);
}

static __device__ __forceinline__ void atomic_max_f32(float* addr, float v) {
    // sign-split trick: int compare == float compare for >=0,
    // uint compare reversed for <0. Works with -1e30f init.
    if (v >= 0.0f) atomicMax((int*)addr, __float_as_int(v));
    else           atomicMin((unsigned int*)addr, __float_as_uint(v));
}

// ---------------- init / cast kernels ----------------
__global__ void fill_f32(float* p, float v, long long n) {
    long long i = (long long)blockIdx.x * blockDim.x + threadIdx.x;
    if (i < n) p[i] = v;
}

// dst[r, 0..Cp) = bf16(src[r, 0..C)), zero-padded K
__global__ void cast_pad_bf16(const float* __restrict__ src, uint16_t* __restrict__ dst,
                              int R, int C, int Cp) {
    long long i = (long long)blockIdx.x * blockDim.x + threadIdx.x;
    if (i >= (long long)R * Cp) return;
    int r = (int)(i / Cp), c = (int)(i % Cp);
    dst[i] = (c < C) ? f32_to_bf16(src[(size_t)r * C + c]) : (uint16_t)0;
}

// BT[n, k] = bf16(W[k, n]); zero pad k >= K. BT is [N x Kp] row-major.
__global__ void cast_transpose_bf16(const float* __restrict__ W, uint16_t* __restrict__ BT,
                                    int K, int N, int Kp) {
    long long i = (long long)blockIdx.x * blockDim.x + threadIdx.x;
    if (i >= (long long)N * Kp) return;
    int n = (int)(i / Kp), k = (int)(i % Kp);
    BT[i] = (k < K) ? f32_to_bf16(W[(size_t)k * N + n]) : (uint16_t)0;
}

// ---------------- WMMA GEMM: C[M,N] = A[M,K] * BT[N,K]^T  (+bias, relu) ----
// Block tile 128(M) x 64(N), 256 threads = 8 waves in 4x2 grid,
// each wave owns a 32x32 sub-tile = 2x2 v_wmma_f32_16x16x32_bf16 tiles.
// Tile staging: CDNA5 async global->LDS DMA (ASYNCcnt), double-buffered so
// the DMA of tile k+1 overlaps the WMMA work on tile k.
#define LDS_STRIDE 40  // 32 + 8 pad (uint16 units); row pitch 80B (16B aligned)

template<int NN, int KK, bool RELU, bool HAS_BIAS, bool WF32, bool WBF16>
__global__ __launch_bounds__(256)
void wmma_gemm_bf16(const uint16_t* __restrict__ A,   // [M x KK] row-major bf16
                    const uint16_t* __restrict__ BT,  // [NN x KK] row-major bf16
                    const float* __restrict__ bias,   // [NN] or null
                    float* __restrict__ Cf,           // [M x NN] f32 out or null
                    uint16_t* __restrict__ Cbf,       // [M x NN] bf16 out or null
                    int M) {
    __shared__ uint16_t As[2][128 * LDS_STRIDE];
    __shared__ uint16_t Bs[2][64 * LDS_STRIDE];

    const int tid  = threadIdx.x;
    const int lane = tid & 31;
    const int wave = tid >> 5;
    const int wrow = wave >> 1;      // 0..3  -> M offset wrow*32
    const int wcol = wave & 1;       // 0..1  -> N offset wcol*32
    const int half = lane >> 4;      // lane group (ISA frag layout)
    const int l16  = lane & 15;
    const int bm = blockIdx.y * 128;
    const int bn = blockIdx.x * 64;

    v8f acc[2][2];
    #pragma unroll
    for (int a = 0; a < 2; ++a)
        #pragma unroll
        for (int b = 0; b < 2; ++b)
            #pragma unroll
            for (int r = 0; r < 8; ++r) acc[a][b][r] = 0.0f;

    // global->LDS async staging assignments:
    // A tile 128x32: thread stages 16 bf16 (2 x b128), B tile 64x32: 8 bf16 (1 x b128)
    const int arow = tid >> 1, akc = (tid & 1) * 16;
    const int brow = tid >> 2, bkc = (tid & 3) * 8;
    int garow = bm + arow;
    if (garow >= M) garow = M - 1;   // clamp: rows >= M only feed masked outputs

    // addrspace(3) byte offsets for the async-load LDS destination VGPR
    const uint32_t lA0 = (uint32_t)(uintptr_t)(&As[0][arow * LDS_STRIDE + akc]);
    const uint32_t lA1 = (uint32_t)(uintptr_t)(&As[1][arow * LDS_STRIDE + akc]);
    const uint32_t lB0 = (uint32_t)(uintptr_t)(&Bs[0][brow * LDS_STRIDE + bkc]);
    const uint32_t lB1 = (uint32_t)(uintptr_t)(&Bs[1][brow * LDS_STRIDE + bkc]);
    const uint16_t* gAbase = A  + (size_t)garow * KK + akc;
    const uint16_t* gBbase = BT + (size_t)(bn + brow) * KK + bkc;

    // GLOBAL_LOAD_ASYNC_TO_LDS_B128: copies 16 bytes per lane from global memory to LDS
    // (offset applies to both sides, ISA 15.18.3); tracked by ASYNCcnt.
    auto issue = [&](uint32_t la, uint32_t lb, int k0) {
        const uint16_t* gA = gAbase + k0;
        const uint16_t* gB = gBbase + k0;
        asm volatile(
            "global_load_async_to_lds_b128 %0, %2, off\n\t"
            "global_load_async_to_lds_b128 %0, %2, off offset:16\n\t"
            "global_load_async_to_lds_b128 %1, %3, off"
            :
            : "v"(la), "v"(lb), "v"(gA), "v"(gB)
            : "memory");
    };

    // prologue: stage tile 0 into buffer 0
    issue(lA0, lB0, 0);
    asm volatile("s_wait_asynccnt 0x0" ::: "memory");
    __syncthreads();

    // pipelined main loop: DMA tile k+1 into the alternate buffer while the
    // matrix pipe consumes tile k; one wait + one barrier per iteration.
    #pragma unroll
    for (int k0 = 0; k0 < KK; k0 += 32) {
        const int cur = (k0 >> 5) & 1;
        if (k0 + 32 < KK)
            issue(cur ? lA0 : lA1, cur ? lB0 : lB1, k0 + 32);  // prefetch next tile

        const uint16_t* __restrict__ Ab = &As[cur][0];
        const uint16_t* __restrict__ Bb = &Bs[cur][0];

        // A fragment (16x32 bf16, ISA 7.12.2): lane<16 holds K 0..7,16..23;
        // lane>=16 holds K 8..15,24..31; 2 packed bf16 per VGPR.
        union Frag { v16bf v; uint32_t u[8]; };
        Frag fa[2], fb[2];
        #pragma unroll
        for (int tm = 0; tm < 2; ++tm) {
            const uint16_t* ap = Ab + (wrow * 32 + tm * 16 + l16) * LDS_STRIDE;
            #pragma unroll
            for (int p = 0; p < 8; ++p) {
                int kk = ((p < 4) ? (2 * p) : (16 + 2 * (p - 4))) + half * 8;
                fa[tm].u[p] = *(const uint32_t*)(ap + kk);
            }
        }
        // B fragment (32x16): lane = column n; lane<16 holds K 0..7,16..23;
        // lane>=16 holds K 8..15,24..31, 2 consecutive K packed per VGPR.
        #pragma unroll
        for (int tn = 0; tn < 2; ++tn) {
            const uint16_t* bp = Bb + (wcol * 32 + tn * 16 + l16) * LDS_STRIDE;
            #pragma unroll
            for (int p = 0; p < 8; ++p) {
                int kk = ((p < 4) ? (2 * p) : (16 + 2 * (p - 4))) + half * 8;
                fb[tn].u[p] = *(const uint32_t*)(bp + kk);
            }
        }

        #pragma unroll
        for (int tm = 0; tm < 2; ++tm)
            #pragma unroll
            for (int tn = 0; tn < 2; ++tn)
                acc[tm][tn] = __builtin_amdgcn_wmma_f32_16x16x32_bf16(
                    false, fa[tm].v, false, fb[tn].v,
                    (short)0, acc[tm][tn], false, false);

        if (k0 + 32 < KK)
            asm volatile("s_wait_asynccnt 0x0" ::: "memory");  // next tile landed
        __syncthreads();   // all waves done reading buf[cur]; next tile visible
    }

    // epilogue: D layout -> lane holds col n=l16, VGPR r holds row r + 8*half
    #pragma unroll
    for (int tm = 0; tm < 2; ++tm) {
        #pragma unroll
        for (int tn = 0; tn < 2; ++tn) {
            const int gn = bn + wcol * 32 + tn * 16 + l16;
            const float bi = HAS_BIAS ? bias[gn] : 0.0f;
            #pragma unroll
            for (int r = 0; r < 8; ++r) {
                const int gm = bm + wrow * 32 + tm * 16 + half * 8 + r;
                if (gm < M) {
                    float v = acc[tm][tn][r] + bi;
                    if (RELU) v = fmaxf(v, 0.0f);
                    if (WF32)  Cf[(size_t)gm * NN + gn]  = v;
                    if (WBF16) Cbf[(size_t)gm * NN + gn] = f32_to_bf16(v);
                }
            }
        }
    }
}

// ---------------- attention scalar kernels ----------------
// a_src[n,h] = sum_c t[n,h,c]*att_src[h,c] ; same for a_dst
__global__ void attn_scores(const float* __restrict__ t,
                            const float* __restrict__ att_src,
                            const float* __restrict__ att_dst,
                            float* __restrict__ asrc, float* __restrict__ adst, int C) {
    int i = blockIdx.x * blockDim.x + threadIdx.x;   // n*HEADS + h
    if (i >= N_NODES * HEADS) return;
    int n = i / HEADS, h = i % HEADS;
    const float* tp = t + (size_t)n * HEADS * C + h * C;
    const float* ws = att_src + h * C;
    const float* wd = att_dst + h * C;
    float s = 0.0f, d = 0.0f;
    for (int c = 0; c < C; ++c) { s += tp[c] * ws[c]; d += tp[c] * wd[c]; }
    asrc[i] = s; adst[i] = d;
}

static __device__ __forceinline__ void edge_sd(const int* __restrict__ ei, int e,
                                               int& s, int& d) {
    if (e < N_EDGES) { s = ei[e]; d = ei[N_EDGES + e]; }
    else             { s = d = e - N_EDGES; }          // self loop
}

// pass 1: leaky-relu logits -> ebuf, segment max into m
__global__ void edge_max(const int* __restrict__ ei, const float* __restrict__ asrc,
                         const float* __restrict__ adst, float* __restrict__ ebuf,
                         float* __restrict__ m) {
    int e = blockIdx.x * blockDim.x + threadIdx.x;
    if (e >= E_TOT) return;
    int s, d; edge_sd(ei, e, s, d);
    #pragma unroll
    for (int h = 0; h < HEADS; ++h) {
        float v = asrc[s * HEADS + h] + adst[d * HEADS + h];
        v = (v >= 0.0f) ? v : NEG_SLOPE * v;
        ebuf[(size_t)e * HEADS + h] = v;
        atomic_max_f32(&m[d * HEADS + h], v);
    }
}

// pass 2: ebuf = exp(ebuf - m[dst]); segment sum into denom
__global__ void edge_expsum(const int* __restrict__ ei, const float* __restrict__ m,
                            float* __restrict__ ebuf, float* __restrict__ denom) {
    int e = blockIdx.x * blockDim.x + threadIdx.x;
    if (e >= E_TOT) return;
    int s, d; edge_sd(ei, e, s, d);
    #pragma unroll
    for (int h = 0; h < HEADS; ++h) {
        float ex = __expf(ebuf[(size_t)e * HEADS + h] - m[d * HEADS + h]);
        ebuf[(size_t)e * HEADS + h] = ex;
        atomicAdd(&denom[d * HEADS + h], ex);
    }
}

// pass 3: acc[dst] += t[src] * alpha ; one thread per (edge, 4 channels)
__global__ void edge_scatter(const int* __restrict__ ei, const float* __restrict__ t,
                             const float* __restrict__ ebuf, const float* __restrict__ denom,
                             float* __restrict__ acc, int HC, int C) {
    long long i = (long long)blockIdx.x * blockDim.x + threadIdx.x;
    int vec = HC / 4;
    if (i >= (long long)E_TOT * vec) return;
    int e = (int)(i / vec), q = (int)(i % vec);
    int s, d; edge_sd(ei, e, s, d);
    int h = (q * 4) / C;
    float alpha = ebuf[(size_t)e * HEADS + h] / (denom[d * HEADS + h] + 1e-16f);
    const float4 tv = *(const float4*)(t + (size_t)s * HC + q * 4);
    float* ap = acc + (size_t)d * HC + q * 4;
    atomicAdd(ap + 0, tv.x * alpha);
    atomicAdd(ap + 1, tv.y * alpha);
    atomicAdd(ap + 2, tv.z * alpha);
    atomicAdd(ap + 3, tv.w * alpha);
}

// out = relu(acc + bias); optionally bf16 copy and/or f32 copy
__global__ void bias_relu_cast(const float* __restrict__ acc, const float* __restrict__ bias,
                               uint16_t* __restrict__ outb, float* __restrict__ outf, int HC) {
    long long i = (long long)blockIdx.x * blockDim.x + threadIdx.x;
    if (i >= (long long)N_NODES * HC) return;
    int f = (int)(i % HC);
    float v = fmaxf(acc[i] + bias[f], 0.0f);
    if (outb) outb[i] = f32_to_bf16(v);
    if (outf) outf[i] = v;
}

// ---------------- pooling + head ----------------
__global__ void pool_scatter(const float* __restrict__ h2, const int* __restrict__ batch,
                             float* __restrict__ pooled, float* __restrict__ counts) {
    long long i = (long long)blockIdx.x * blockDim.x + threadIdx.x;
    if (i >= (long long)N_NODES * OUTD) return;
    int n = (int)(i / OUTD), f = (int)(i % OUTD);
    int g = batch[n];
    atomicAdd(&pooled[g * OUTD + f], h2[i]);
    if (f == 0) atomicAdd(&counts[g], 1.0f);
}

__global__ void head_kernel(const float* __restrict__ pooled, const float* __restrict__ counts,
                            const float* __restrict__ Wc, const float* __restrict__ bc,
                            const float* __restrict__ Wf, const float* __restrict__ bf,
                            float* __restrict__ out) {
    int g = threadIdx.x;
    if (g >= N_GRAPHS) return;
    float inv = 1.0f / fmaxf(counts[g], 1.0f);
    float l0 = bc[0], l1 = bc[1], cf = bf[0];
    for (int f = 0; f < OUTD; ++f) {
        float p = pooled[g * OUTD + f] * inv;
        l0 += p * Wc[f * 2 + 0];
        l1 += p * Wc[f * 2 + 1];
        cf += p * Wf[f];
    }
    out[g * 2 + 0] = l0;
    out[g * 2 + 1] = l1;
    out[2 * N_GRAPHS + g] = 1.0f / (1.0f + __expf(-cf));   // confidence after logits block
}

// ---------------- host orchestration ----------------
extern "C" void kernel_launch(void* const* d_in, const int* in_sizes, int n_in,
                              void* d_out, int out_size, void* d_ws, size_t ws_size,
                              hipStream_t stream) {
    const float* x      = (const float*)d_in[0];
    const int*   ei     = (const int*)  d_in[1];
    const int*   batch  = (const int*)  d_in[2];
    const float* W_in   = (const float*)d_in[3];
    const float* b_in   = (const float*)d_in[4];
    const float* W1     = (const float*)d_in[5];
    const float* as1    = (const float*)d_in[6];
    const float* ad1    = (const float*)d_in[7];
    const float* bias1  = (const float*)d_in[8];
    const float* W2     = (const float*)d_in[9];
    const float* as2    = (const float*)d_in[10];
    const float* ad2    = (const float*)d_in[11];
    const float* bias2  = (const float*)d_in[12];
    const float* Wcls   = (const float*)d_in[13];
    const float* bcls   = (const float*)d_in[14];
    const float* Wconf  = (const float*)d_in[15];
    const float* bconf  = (const float*)d_in[16];

    // carve workspace (bump allocator, 256B aligned); total ~225 MB
    char* ws = (char*)d_ws;
    size_t off = 0;
    auto carve = [&](size_t bytes) -> void* {
        void* p = ws + off;
        off = (off + bytes + 255) & ~(size_t)255;
        return p;
    };
    uint16_t* xb    = (uint16_t*)carve((size_t)N_NODES * KP_IN * 2);  // x in bf16, K padded
    uint16_t* wb    = (uint16_t*)carve((size_t)HID * KP_IN * 2);      // B^T scratch (max)
    uint16_t* hb    = (uint16_t*)carve((size_t)N_NODES * HID * 2);    // activations bf16
    float*    t     = (float*)   carve((size_t)N_NODES * HID * 4);    // GEMM f32 out
    float*    acc   = (float*)   carve((size_t)N_NODES * HID * 4);    // conv accumulators
    float*    asrc  = (float*)   carve((size_t)N_NODES * HEADS * 4);
    float*    adst  = (float*)   carve((size_t)N_NODES * HEADS * 4);
    float*    segm  = (float*)   carve((size_t)N_NODES * HEADS * 4);
    float*    den   = (float*)   carve((size_t)N_NODES * HEADS * 4);
    float*    ebuf  = (float*)   carve((size_t)E_TOT * HEADS * 4);
    float*    pooled= (float*)   carve((size_t)N_GRAPHS * OUTD * 4);
    float*    counts= (float*)   carve((size_t)N_GRAPHS * 4);

    const int T = 256;
    auto blk1d = [](long long n, int t) { return (int)((n + t - 1) / t); };
    const dim3 gridM256(HID / 64, (N_NODES + 127) / 128);
    const dim3 gridM128(OUTD / 64, (N_NODES + 127) / 128);

    // ---- input projection: h0 = relu(x @ W_in + b_in), bf16 out ----
    cast_pad_bf16<<<blk1d((long long)N_NODES * KP_IN, T), T, 0, stream>>>(x, xb, N_NODES, IN_DIM, KP_IN);
    cast_transpose_bf16<<<blk1d((long long)HID * KP_IN, T), T, 0, stream>>>(W_in, wb, IN_DIM, HID, KP_IN);
    wmma_gemm_bf16<HID, KP_IN, true, true, false, true>
        <<<gridM256, 256, 0, stream>>>(xb, wb, b_in, nullptr, hb, N_NODES);

    // ---- GAT conv 1 (HID -> 4 heads x 64) ----
    cast_transpose_bf16<<<blk1d((long long)HID * HID, T), T, 0, stream>>>(W1, wb, HID, HID, HID);
    wmma_gemm_bf16<HID, HID, false, false, true, false>
        <<<gridM256, 256, 0, stream>>>(hb, wb, nullptr, t, nullptr, N_NODES);
    attn_scores<<<blk1d((long long)N_NODES * HEADS, T), T, 0, stream>>>(t, as1, ad1, asrc, adst, HID / HEADS);
    fill_f32<<<blk1d((long long)N_NODES * HEADS, T), T, 0, stream>>>(segm, -1e30f, (long long)N_NODES * HEADS);
    fill_f32<<<blk1d((long long)N_NODES * HEADS, T), T, 0, stream>>>(den, 0.0f, (long long)N_NODES * HEADS);
    fill_f32<<<blk1d((long long)N_NODES * HID, T), T, 0, stream>>>(acc, 0.0f, (long long)N_NODES * HID);
    edge_max<<<blk1d(E_TOT, T), T, 0, stream>>>(ei, asrc, adst, ebuf, segm);
    edge_expsum<<<blk1d(E_TOT, T), T, 0, stream>>>(ei, segm, ebuf, den);
    edge_scatter<<<blk1d((long long)E_TOT * (HID / 4), T), T, 0, stream>>>(ei, t, ebuf, den, acc, HID, HID / HEADS);
    bias_relu_cast<<<blk1d((long long)N_NODES * HID, T), T, 0, stream>>>(acc, bias1, hb, nullptr, HID);

    // ---- GAT conv 2 (HID -> 4 heads x 32) ----
    cast_transpose_bf16<<<blk1d((long long)OUTD * HID, T), T, 0, stream>>>(W2, wb, HID, OUTD, HID);
    wmma_gemm_bf16<OUTD, HID, false, false, true, false>
        <<<gridM128, 256, 0, stream>>>(hb, wb, nullptr, t, nullptr, N_NODES);
    attn_scores<<<blk1d((long long)N_NODES * HEADS, T), T, 0, stream>>>(t, as2, ad2, asrc, adst, OUTD / HEADS);
    fill_f32<<<blk1d((long long)N_NODES * HEADS, T), T, 0, stream>>>(segm, -1e30f, (long long)N_NODES * HEADS);
    fill_f32<<<blk1d((long long)N_NODES * HEADS, T), T, 0, stream>>>(den, 0.0f, (long long)N_NODES * HEADS);
    fill_f32<<<blk1d((long long)N_NODES * OUTD, T), T, 0, stream>>>(acc, 0.0f, (long long)N_NODES * OUTD);
    edge_max<<<blk1d(E_TOT, T), T, 0, stream>>>(ei, asrc, adst, ebuf, segm);
    edge_expsum<<<blk1d(E_TOT, T), T, 0, stream>>>(ei, segm, ebuf, den);
    edge_scatter<<<blk1d((long long)E_TOT * (OUTD / 4), T), T, 0, stream>>>(ei, t, ebuf, den, acc, OUTD, OUTD / HEADS);
    // final conv output (f32) reuses t
    bias_relu_cast<<<blk1d((long long)N_NODES * OUTD, T), T, 0, stream>>>(acc, bias2, nullptr, t, OUTD);

    // ---- global mean pool + classifier head ----
    fill_f32<<<blk1d((long long)N_GRAPHS * OUTD, T), T, 0, stream>>>(pooled, 0.0f, (long long)N_GRAPHS * OUTD);
    fill_f32<<<1, N_GRAPHS, 0, stream>>>(counts, 0.0f, N_GRAPHS);
    pool_scatter<<<blk1d((long long)N_NODES * OUTD, T), T, 0, stream>>>(t, batch, pooled, counts);
    head_kernel<<<1, N_GRAPHS, 0, stream>>>(pooled, counts, Wcls, bcls, Wconf, bconf, (float*)d_out);
}
